// Eventformer_39986145526292
// MI455X (gfx1250) — matrix operand order repaired and checked by reading, not compiled, not run on hardware
//
#include <hip/hip_runtime.h>
#include <hip/hip_bf16.h>
#include <math.h>

typedef __attribute__((ext_vector_type(16))) __bf16 bf16x16;
typedef __attribute__((ext_vector_type(8)))  float  f32x8;
typedef unsigned short u16;

#define WAVE 32

// Fragment viewed as 8 packed dwords (2 bf16 each)
union BF16Frag { bf16x16 v; unsigned u[8]; };

// Pack two f32 into one dword of bf16 pair (x -> low half, y -> high half).
__device__ __forceinline__ unsigned cvt_pk(float x, float y) {
#if __has_builtin(__builtin_amdgcn_cvt_pk_bf16_f32)
  typedef __attribute__((ext_vector_type(2))) __bf16 bf16x2;
  union { bf16x2 v; unsigned u; } c;
  c.v = __builtin_amdgcn_cvt_pk_bf16_f32(x, y);
  return c.u;
#else
  union { float f; unsigned u; } a, b;
  a.f = x; b.f = y;
  unsigned au = a.u + 0x8000u;   // round-half-up to bf16
  unsigned bu = b.u + 0x8000u;
  return __builtin_amdgcn_perm(bu, au, 0x07060302u);  // {bu[31:16], au[31:16]}
#endif
}

__device__ __forceinline__ void packpair4(unsigned* up, float4 u, float4 v) {
  up[0] = cvt_pk(u.x, u.y);
  up[1] = cvt_pk(u.z, u.w);
  up[2] = cvt_pk(v.x, v.y);
  up[3] = cvt_pk(v.z, v.w);
}

__device__ __forceinline__ u16 f2bfu(float x) {
  union { float f; unsigned u; } c; c.f = x;
  return (u16)((c.u + 0x8000u) >> 16);
}

__device__ __forceinline__ float bf2f(u16 h) {
  union { unsigned u; float f; } c; c.u = ((unsigned)h) << 16;
  return c.f;
}

__device__ __forceinline__ float gelu_f(float x) {
  return 0.5f * x * (1.0f + erff(x * 0.7071067811865476f));
}

// load 16 contiguous bf16 (two uint4) into a fragment
__device__ __forceinline__ void load_frag16(BF16Frag& f, const u16* p0, const u16* p1) {
  uint4 lo = *(const uint4*)p0;
  uint4 hi = *(const uint4*)p1;
  f.u[0] = lo.x; f.u[1] = lo.y; f.u[2] = lo.z; f.u[3] = lo.w;
  f.u[4] = hi.x; f.u[5] = hi.y; f.u[6] = hi.z; f.u[7] = hi.w;
}

// ---------------------------------------------------------------------------
// WMMA GEMM, bf16 in / f32 acc: Y[M,N] = act(X[M,K] @ Wt^T + bias) (+res)
// X: bf16 [M,K]; Wt: bf16 TRANSPOSED [N,K]. M%64==0, N%16==0, K%32==0.
// One wave = 64x16 strip. Per K-step: 2 b128 (B) + 8 b128 (A) + 4 WMMAs.
// obf: 0 -> Y float (res optional), 1 -> Y bf16 (no res).
// ---------------------------------------------------------------------------
__global__ void k_gemm(const u16* __restrict__ X, const u16* __restrict__ Wt,
                       const float* __restrict__ bias, const float* __restrict__ res,
                       void* __restrict__ Yv, int M, int N, int K, int act, int obf) {
  const int lane = threadIdx.x;
  const int row0 = blockIdx.y * 64;
  const int col0 = blockIdx.x * 16;
  const int mloc = lane & 15;
  const int half = lane >> 4;
  const int ncol = col0 + mloc;
  f32x8 acc0 = {0.f,0.f,0.f,0.f,0.f,0.f,0.f,0.f};
  f32x8 acc1 = acc0, acc2 = acc0, acc3 = acc0;

  const u16* wrow = Wt + (size_t)ncol * K;
  for (int k0 = 0; k0 < K; k0 += 32) {
    // B fragment: lanes 0-15 K=k0..k0+15, lanes 16-31 K=k0+16..k0+31 (contiguous)
    BF16Frag bfr;
    const int kb = k0 + half * 16;
    load_frag16(bfr, wrow + kb, wrow + kb + 8);
    if (k0 + 32 < K) __builtin_prefetch(wrow + kb + 32, 0, 3);

    // A fragments: elements 0..15 -> K=ka..ka+15 (contiguous)
    const int ka = k0 + half * 16;
#pragma unroll
    for (int t = 0; t < 4; ++t) {
      const u16* xr = X + (size_t)(row0 + t * 16 + mloc) * K + ka;
      BF16Frag a;
      load_frag16(a, xr, xr + 8);
      if (t == 0)
        acc0 = __builtin_amdgcn_wmma_f32_16x16x32_bf16(false, a.v, false, bfr.v, (short)0, acc0, false, false);
      else if (t == 1)
        acc1 = __builtin_amdgcn_wmma_f32_16x16x32_bf16(false, a.v, false, bfr.v, (short)0, acc1, false, false);
      else if (t == 2)
        acc2 = __builtin_amdgcn_wmma_f32_16x16x32_bf16(false, a.v, false, bfr.v, (short)0, acc2, false, false);
      else
        acc3 = __builtin_amdgcn_wmma_f32_16x16x32_bf16(false, a.v, false, bfr.v, (short)0, acc3, false, false);
    }
  }

  const float bv = bias ? bias[ncol] : 0.f;
#pragma unroll
  for (int t = 0; t < 4; ++t) {
    const f32x8& acc = (t == 0) ? acc0 : (t == 1) ? acc1 : (t == 2) ? acc2 : acc3;
    const int rbase = row0 + t * 16 + half * 8;   // C/D: vgpr r -> row, lane&15 -> col
#pragma unroll
    for (int r = 0; r < 8; ++r) {
      const size_t off = (size_t)(rbase + r) * N + ncol;
      float v = acc[r] + bv;
      if (act == 1) v = gelu_f(v);
      if (obf) {
        ((u16*)Yv)[off] = f2bfu(v);
      } else {
        if (res) v += res[off];
        ((float*)Yv)[off] = v;
      }
    }
  }
}

// Weight convert + transpose: Wt[n*K+k] = bf16(W[k*N+n])
__global__ void k_wcvt(const float* __restrict__ W, u16* __restrict__ Wt, int K, int N) {
  const int id = blockIdx.x * blockDim.x + threadIdx.x;
  if (id < K * N) {
    const int n = id / K, k = id % K;
    Wt[id] = f2bfu(W[(size_t)k * N + n]);
  }
}

// Elementwise f32 -> bf16
__global__ void k_cvt(const float* __restrict__ X, u16* __restrict__ Y, size_t n) {
  size_t i = (size_t)blockIdx.x * blockDim.x + threadIdx.x;
  if (i < n) Y[i] = f2bfu(X[i]);
}

// ---------------------------------------------------------------------------
// Row layernorm (wave32 shuffle reductions). obf: 0 -> f32 out, 1 -> bf16 out.
// ---------------------------------------------------------------------------
__global__ void k_layernorm(const float* __restrict__ X, const float* __restrict__ g,
                            const float* __restrict__ be, void* __restrict__ Yv,
                            int d, int obf) {
  const int row = blockIdx.x;
  const int lane = threadIdx.x;
  const float* xr = X + (size_t)row * d;
  float s = 0.f;
  for (int c = lane; c < d; c += WAVE) s += xr[c];
  for (int m = 16; m >= 1; m >>= 1) s += __shfl_xor(s, m, WAVE);
  float mu = s / (float)d;
  float v = 0.f;
  for (int c = lane; c < d; c += WAVE) { float t = xr[c] - mu; v += t * t; }
  for (int m = 16; m >= 1; m >>= 1) v += __shfl_xor(v, m, WAVE);
  float inv = rsqrtf(v / (float)d + 1e-5f);
  if (obf) {
    u16* yr = (u16*)Yv + (size_t)row * d;
    for (int c = lane; c < d; c += WAVE) yr[c] = f2bfu((xr[c] - mu) * inv * g[c] + be[c]);
  } else {
    float* yr = (float*)Yv + (size_t)row * d;
    for (int c = lane; c < d; c += WAVE) yr[c] = (xr[c] - mu) * inv * g[c] + be[c];
  }
}

// ---------------------------------------------------------------------------
// PAAA global attention, flash-style (dh = 32, Br = 16, Bc = 32), all-WMMA.
// Q,K,V bf16; O bf16. grid (Ntok/16, H, B), block 32. Ntok % 32 == 0.
// ---------------------------------------------------------------------------
__global__ void k_paaa_flash(const u16* __restrict__ Q, const u16* __restrict__ K,
                             const u16* __restrict__ V, const float* __restrict__ pol,
                             const float* __restrict__ polbias, u16* __restrict__ O,
                             int Ntok, int D, int H) {
  __shared__ __align__(16) float sS[16 * 32];
  __shared__ float sM[16], sL[16], sA[16];
  __shared__ float sPol[32];
  const int lane = threadIdx.x;
  const int b = blockIdx.z, h = blockIdx.y;
  const int row0 = blockIdx.x * 16;
  const size_t base = ((size_t)b * Ntok) * D + (size_t)h * 32;
  const int mloc = lane & 15;
  const int half = lane >> 4;
  const float scale = 0.17677669529663687f;  // 1/sqrt(32)

  if (lane < 16) { sM[lane] = -1e30f; sL[lane] = 0.f; sA[lane] = 1.f; }

  float bias0 = 0.f, bias1 = 0.f;
  if (lane < 16) {
    int piidx = pol[(size_t)b * Ntok + row0 + lane] > 0.f ? 1 : 0;
    bias0 = polbias[h * 4 + piidx * 2 + 0];
    bias1 = polbias[h * 4 + piidx * 2 + 1];
  }

  BF16Frag aq;
  {
    const u16* qr = Q + base + (size_t)(row0 + mloc) * D + half * 16;
    load_frag16(aq, qr, qr + 8);
  }
  f32x8 acc0 = {0.f,0.f,0.f,0.f,0.f,0.f,0.f,0.f};
  f32x8 acc1 = acc0;
  const f32x8 zc = acc0;
  const int kb = half * 16;
  __syncthreads();

  for (int j0 = 0; j0 < Ntok; j0 += 32) {
    // S = Q (16x32) x K_tile^T : two WMMAs; key feature rows contiguous -> b128
    BF16Frag bk;
    {
      const u16* kr = K + base + (size_t)(j0 + mloc) * D + kb;
      load_frag16(bk, kr, kr + 8);
    }
    f32x8 s0 = __builtin_amdgcn_wmma_f32_16x16x32_bf16(false, aq.v, false, bk.v, (short)0, zc, false, false);
    {
      const u16* kr = K + base + (size_t)(j0 + 16 + mloc) * D + kb;
      load_frag16(bk, kr, kr + 8);
    }
    f32x8 s1 = __builtin_amdgcn_wmma_f32_16x16x32_bf16(false, aq.v, false, bk.v, (short)0, zc, false, false);
    sPol[lane] = pol[(size_t)b * Ntok + j0 + lane];
#pragma unroll
    for (int r = 0; r < 8; ++r) {
      sS[(half * 8 + r) * 32 + mloc]      = s0[r];
      sS[(half * 8 + r) * 32 + 16 + mloc] = s1[r];
    }
    __syncthreads();
    // online softmax per row (lanes 0..15 own rows)
    if (lane < 16) {
      float mo = sM[lane];
      float mx = mo;
      float sv[32];
#pragma unroll
      for (int j = 0; j < 32; ++j) {
        float sval = sS[lane * 32 + j] * scale + (sPol[j] > 0.f ? bias1 : bias0);
        sv[j] = sval;
        mx = fmaxf(mx, sval);
      }
      float alpha = __expf(mo - mx);
      float l = sL[lane] * alpha;
#pragma unroll
      for (int j = 0; j < 32; ++j) {
        float p = __expf(sv[j] - mx);
        sS[lane * 32 + j] = p;
        l += p;
      }
      sM[lane] = mx; sL[lane] = l; sA[lane] = alpha;
    }
    __syncthreads();
#pragma unroll
    for (int r = 0; r < 8; ++r) {
      float al = sA[half * 8 + r];
      acc0[r] *= al; acc1[r] *= al;
    }
    // O += P (16x32) x V_tile (32x32): two WMMAs (output column halves)
    BF16Frag ap;
    {
      const float* pr = &sS[mloc * 32 + half * 16];
      float4 p0 = *(const float4*)(pr);
      float4 p1 = *(const float4*)(pr + 4);
      float4 p2 = *(const float4*)(pr + 8);
      float4 p3 = *(const float4*)(pr + 12);
      packpair4(ap.u, p0, p1);
      packpair4(ap.u + 4, p2, p3);
    }
    BF16Frag bv;
    {
      const u16* vp = V + base + (size_t)(j0 + kb) * D + mloc;
#pragma unroll
      for (int e = 0; e < 16; e += 2)
        bv.u[e >> 1] = (unsigned)vp[(size_t)e * D] | ((unsigned)vp[(size_t)(e + 1) * D] << 16);
    }
    acc0 = __builtin_amdgcn_wmma_f32_16x16x32_bf16(false, ap.v, false, bv.v, (short)0, acc0, false, false);
    {
      const u16* vp = V + base + (size_t)(j0 + kb) * D + 16 + mloc;
#pragma unroll
      for (int e = 0; e < 16; e += 2)
        bv.u[e >> 1] = (unsigned)vp[(size_t)e * D] | ((unsigned)vp[(size_t)(e + 1) * D] << 16);
    }
    acc1 = __builtin_amdgcn_wmma_f32_16x16x32_bf16(false, ap.v, false, bv.v, (short)0, acc1, false, false);
    __syncthreads();
  }
#pragma unroll
  for (int r = 0; r < 8; ++r) {
    int row = half * 8 + r;
    float inv = 1.f / sL[row];
    size_t orow = base + (size_t)(row0 + row) * D;
    O[orow + mloc]      = f2bfu(acc0[r] * inv);
    O[orow + 16 + mloc] = f2bfu(acc1[r] * inv);
  }
}

// ---------------------------------------------------------------------------
// Exact kNN (k=32): 32 rounds of lexicographic (d2bits, idx) min-selection.
// grid (Ntok, B), block 32.
// ---------------------------------------------------------------------------
__global__ void k_knn(const float* __restrict__ coords, const float* __restrict__ times,
                      int Ntok, int* __restrict__ knn_out) {
  const int lane = threadIdx.x;
  const int i = blockIdx.x, b = blockIdx.y;
  const float xi = coords[((size_t)b * Ntok + i) * 2 + 0];
  const float yi = coords[((size_t)b * Ntok + i) * 2 + 1];
  const float ti = times[(size_t)b * Ntok + i];
  unsigned long long prev = 0ull;
  for (int r = 0; r < 32; ++r) {
    unsigned long long best = ~0ull;
    for (int j = lane; j < Ntok; j += WAVE) {
      float dx = coords[((size_t)b * Ntok + j) * 2 + 0] - xi;
      float dy = coords[((size_t)b * Ntok + j) * 2 + 1] - yi;
      float dt = times[(size_t)b * Ntok + j] - ti;
      float d2 = dx * dx + dy * dy + 0.5f * dt * dt;
      unsigned long long key =
          ((unsigned long long)__float_as_uint(d2) << 32) | (unsigned)j;
      if ((r == 0 || key > prev) && key < best) best = key;
    }
    for (int m = 16; m >= 1; m >>= 1) {
      unsigned long long o = __shfl_xor(best, m, WAVE);
      if (o < best) best = o;
    }
    if (lane == 0) knn_out[((size_t)b * Ntok + i) * 32 + r] = (int)(best & 0xFFFFFFFFull);
    prev = best;
  }
}

// ---------------------------------------------------------------------------
// ASNA neighborhood attention over precomputed kNN (kk = 32, dh = 32).
// Q,K,V bf16; O bf16. grid (Ntok, B), block 32.
// ---------------------------------------------------------------------------
__global__ void k_asna_attn(const u16* __restrict__ Q, const u16* __restrict__ K,
                            const u16* __restrict__ V, const int* __restrict__ knn_in,
                            u16* __restrict__ O, int Ntok, int D, int H) {
  __shared__ float sAw[32];
  __shared__ int sIdx[32];
  const int lane = threadIdx.x;
  const int i = blockIdx.x, b = blockIdx.y;
  sIdx[lane] = knn_in[((size_t)b * Ntok + i) * 32 + lane];
  __syncthreads();
  for (int h = 0; h < H; ++h) {
    const u16* qp = Q + ((size_t)b * Ntok + i) * D + h * 32;
    const int nb = sIdx[lane];
    const u16* kp = K + ((size_t)b * Ntok + nb) * D + h * 32;
    float dot = 0.f;
#pragma unroll
    for (int dd = 0; dd < 32; ++dd) dot += bf2f(qp[dd]) * bf2f(kp[dd]);
    dot *= 0.17677669529663687f;
    float mx = dot;
    for (int m = 16; m >= 1; m >>= 1) mx = fmaxf(mx, __shfl_xor(mx, m, WAVE));
    float e = __expf(dot - mx);
    float sum = e;
    for (int m = 16; m >= 1; m >>= 1) sum += __shfl_xor(sum, m, WAVE);
    sAw[lane] = e / sum;
    __syncthreads();
    float o = 0.f;
    for (int kk = 0; kk < 32; ++kk) {
      int nb2 = sIdx[kk];
      o += sAw[kk] * bf2f(V[((size_t)b * Ntok + nb2) * D + h * 32 + lane]);
    }
    O[((size_t)b * Ntok + i) * D + h * 32 + lane] = f2bfu(o);
    __syncthreads();
  }
}

// ---------------------------------------------------------------------------
// Farthest point sampling (inherently serial): one block per batch.
// ---------------------------------------------------------------------------
__global__ void k_fps(const float* __restrict__ coords, const float* __restrict__ times,
                      float* __restrict__ dists, int* __restrict__ outIdx,
                      int Ntok, int M) {
  __shared__ float sv[256];
  __shared__ int si[256];
  __shared__ int sPrev;
  const int b = blockIdx.x, tid = threadIdx.x;
  float* db = dists + (size_t)b * 4096;
  for (int j = tid; j < Ntok; j += 256) db[j] = 1e30f;
  if (tid == 0) { outIdx[(size_t)b * M] = 0; sPrev = 0; }
  __syncthreads();
  for (int m = 1; m < M; ++m) {
    const int p = sPrev;
    const float px = coords[((size_t)b * Ntok + p) * 2 + 0];
    const float py = coords[((size_t)b * Ntok + p) * 2 + 1];
    const float pt = times[(size_t)b * Ntok + p];
    float bv = -1.f; int bix = 0x7FFFFFFF;
    for (int j = tid; j < Ntok; j += 256) {
      float dx = coords[((size_t)b * Ntok + j) * 2 + 0] - px;
      float dy = coords[((size_t)b * Ntok + j) * 2 + 1] - py;
      float dt = times[(size_t)b * Ntok + j] - pt;
      float d = sqrtf(dx * dx + dy * dy + dt * dt + 1e-8f);
      float dm = fminf(db[j], d);
      db[j] = dm;
      if (dm > bv) { bv = dm; bix = j; }
    }
    sv[tid] = bv; si[tid] = bix;
    __syncthreads();
    for (int s = 128; s >= 1; s >>= 1) {
      if (tid < s) {
        if (sv[tid + s] > sv[tid] || (sv[tid + s] == sv[tid] && si[tid + s] < si[tid])) {
          sv[tid] = sv[tid + s]; si[tid] = si[tid + s];
        }
      }
      __syncthreads();
    }
    if (tid == 0) { outIdx[(size_t)b * M + m] = si[0]; sPrev = si[0]; }
    __syncthreads();
  }
}

__global__ void k_gather_feats(const float* __restrict__ X, const int* __restrict__ idx,
                               float* __restrict__ Y, int Ntok, int M, int d) {
  const int b = blockIdx.y, m = blockIdx.x;
  const int src = idx[(size_t)b * M + m];
  const float* xr = X + ((size_t)b * Ntok + src) * d;
  float* yr = Y + ((size_t)b * M + m) * d;
  for (int c = threadIdx.x; c < d; c += blockDim.x) yr[c] = xr[c];
}

__global__ void k_gather_meta(const float* __restrict__ cA, const float* __restrict__ tA,
                              const float* __restrict__ pA, const int* __restrict__ idx,
                              float* __restrict__ cB, float* __restrict__ tB,
                              float* __restrict__ pB, int Ntok, int M) {
  const int b = blockIdx.x;
  for (int m = threadIdx.x; m < M; m += blockDim.x) {
    const int src = idx[(size_t)b * M + m];
    cB[((size_t)b * M + m) * 2 + 0] = cA[((size_t)b * Ntok + src) * 2 + 0];
    cB[((size_t)b * M + m) * 2 + 1] = cA[((size_t)b * Ntok + src) * 2 + 1];
    tB[(size_t)b * M + m] = tA[(size_t)b * Ntok + src];
    pB[(size_t)b * M + m] = pA[(size_t)b * Ntok + src];
  }
}

// concat [pa | ao] -> bf16 (feeds gate GEMM only)
__global__ void k_concat(const float* __restrict__ A, const float* __restrict__ Bm,
                         u16* __restrict__ Y, int rows, int d) {
  size_t i = (size_t)blockIdx.x * blockDim.x + threadIdx.x;
  size_t total = (size_t)rows * 2 * d;
  if (i < total) {
    int r = (int)(i / (2 * d));
    int c = (int)(i % (2 * d));
    Y[i] = f2bfu((c < d) ? A[(size_t)r * d + c] : Bm[(size_t)r * d + (c - d)]);
  }
}

__global__ void k_gate(const float* __restrict__ pa, const float* __restrict__ ao,
                       const float* __restrict__ gl, float* __restrict__ Y, size_t n) {
  size_t i = (size_t)blockIdx.x * blockDim.x + threadIdx.x;
  if (i < n) {
    float g = 1.f / (1.f + __expf(-gl[i]));
    Y[i] = g * pa[i] + (1.f - g) * ao[i];
  }
}

// Embedding feature build -> bf16 (feeds fus1 GEMM only)
__global__ void k_embed_feat(const float* __restrict__ coords, const float* __restrict__ times,
                             const float* __restrict__ pol,
                             const float* __restrict__ ctpe_w, const float* __restrict__ ctpe_b,
                             const float* __restrict__ spe_w, const float* __restrict__ spe_b,
                             const float* __restrict__ poltab, u16* __restrict__ C,
                             int total) {
  const int i = blockIdx.x * blockDim.x + threadIdx.x;
  const int tok = i / 192, j = i % 192;
  if (tok >= total) return;
  float val;
  if (j < 64) {
    const float t = times[tok];
    float acc = ctpe_b[j];
#pragma unroll
    for (int s = 0; s < 8; ++s) {
      float ang = t * (float)(1 << s);
      acc += sinf(ang) * ctpe_w[s * 64 + j] + cosf(ang) * ctpe_w[(8 + s) * 64 + j];
    }
    val = acc;
  } else if (j < 128) {
    const int jj = j - 64;
    const float xs = coords[(size_t)tok * 2 + 0] / 346.0f;
    const float ys = coords[(size_t)tok * 2 + 1] / 260.0f;
    val = xs * spe_w[0 * 64 + jj] + ys * spe_w[1 * 64 + jj] + spe_b[jj];
  } else {
    const int jj = j - 128;
    const int pidx = pol[tok] > 0.f ? 1 : 0;
    val = poltab[pidx * 64 + jj];
  }
  C[(size_t)tok * 192 + j] = f2bfu(val);
}

// ---------------------------------------------------------------------------
static inline void gemm(hipStream_t st, const u16* X, const u16* Wt, const float* b,
                        const float* res, void* Y, int M, int N, int K, int act, int obf) {
  dim3 grid(N / 16, M / 64);
  k_gemm<<<grid, 32, 0, st>>>(X, Wt, b, res, Y, M, N, K, act, obf);
}

extern "C" void kernel_launch(void* const* d_in, const int* in_sizes, int n_in,
                              void* d_out, int out_size, void* d_ws, size_t ws_size,
                              hipStream_t stream) {
  (void)in_sizes; (void)n_in; (void)out_size; (void)ws_size;
  const int Bb = 2, N0 = 4096;
  const int dims[4]   = {64, 128, 256, 512};
  const int headsA[4] = {2, 4, 8, 16};
  const int depths[4] = {2, 2, 6, 2};
  const int blkDim[12] = {64,64,128,128,256,256,256,256,256,256,512,512};

  const float* inCoords = (const float*)d_in[0];
  const float* inTimes  = (const float*)d_in[1];
  const float* inPol    = (const float*)d_in[2];

  // ---- param unpack: JAX pytree order (sorted dict keys), after 3 inputs ----
  int pidx = 3;
  auto np = [&]() { return (const float*)d_in[pidx++]; };
  struct Lin { const float *b, *w; };
  struct Ln  { const float *b, *g; };
  struct DownP { Ln ln; Lin proj; } down[3];
  for (int i = 0; i < 3; ++i) {
    down[i].ln.b = np(); down[i].ln.g = np();
    down[i].proj.b = np(); down[i].proj.w = np();
  }
  Lin ctpe; ctpe.b = np(); ctpe.w = np();
  Lin fus1; fus1.b = np(); fus1.w = np();
  Lin fus2; fus2.b = np(); fus2.w = np();
  Ln fus_ln; fus_ln.b = np(); fus_ln.g = np();
  const float* poltab = np();
  Lin spe; spe.b = np(); spe.w = np();

  struct Blk {
    Lin ak; Ln aln1, aln2; Lin am1, am2, aoL, aq, av;
    Lin ffn1, ffn2, gate; Ln n1, n2;
    Lin pk, po; const float* polbias; Lin pq, pv;
  } blk[12];
  for (int t = 0; t < 12; ++t) {
    Blk& Bl = blk[t];
    Bl.ak.b = np(); Bl.ak.w = np();
    Bl.aln1.b = np(); Bl.aln1.g = np();
    Bl.aln2.b = np(); Bl.aln2.g = np();
    Bl.am1.b = np(); Bl.am1.w = np();
    Bl.am2.b = np(); Bl.am2.w = np();
    Bl.aoL.b = np(); Bl.aoL.w = np();
    Bl.aq.b = np(); Bl.aq.w = np();
    Bl.av.b = np(); Bl.av.w = np();
    Bl.ffn1.b = np(); Bl.ffn1.w = np();
    Bl.ffn2.b = np(); Bl.ffn2.w = np();
    Bl.gate.b = np(); Bl.gate.w = np();
    Bl.n1.b = np(); Bl.n1.g = np();
    Bl.n2.b = np(); Bl.n2.g = np();
    Bl.pk.b = np(); Bl.pk.w = np();
    Bl.po.b = np(); Bl.po.w = np();
    Bl.polbias = np();
    Bl.pq.b = np(); Bl.pq.w = np();
    Bl.pv.b = np(); Bl.pv.w = np();
  }

  // ---- workspace layout ----
  float* w = (float*)d_ws;
  const size_t TB = (size_t)1 << 20;                 // 1M elements per token buffer
  float* xb = w; w += TB;                            // f32 residual / feature stream
  float* pa = w; w += TB;
  float* ao = w; w += TB;
  float* t1 = w; w += TB;                            // f32 scratch (gate lin, proj out)
  float* t2 = w; w += TB;                            // f32 scratch (gather staging)
  float* cA = w; w += 2 * 4096 * 2;
  float* cB = w; w += 2 * 4096 * 2;
  float* tA = w; w += 2 * 4096;
  float* tBv = w; w += 2 * 4096;
  float* pA = w; w += 2 * 4096;
  float* pB = w; w += 2 * 4096;
  float* fpsD = w; w += 2 * 4096;
  int* fpsI = (int*)w; w += 2 * 1024;
  int* knnBuf = (int*)w; w += 2 * 4096 * 32;
  // bf16 region (2-byte elements; all counts keep 16B alignment)
  u16* t1h = (u16*)w;
  u16* t2h = t1h + TB;
  u16* qh  = t2h + TB;
  u16* kh  = qh + TB;
  u16* vh  = kh + TB;
  u16* hbh = vh + TB;                                // 4M bf16 (hidden / concat / embed)
  u16* warena = hbh + ((size_t)4 << 20);             // bf16 transposed weights (~25.2M)
  size_t woff = 0;
  auto cvtW = [&](const float* src, int K, int N) -> const u16* {
    u16* dst = warena + woff;
    woff += (size_t)K * N;
    const int total = K * N;
    k_wcvt<<<(total + 255) / 256, 256, 0, stream>>>(src, dst, K, N);
    return dst;
  };

  // one-time (per launch) weight convert+transpose into bf16 arena
  const u16 *fus1w = cvtW(fus1.w, 192, 128), *fus2w = cvtW(fus2.w, 128, 64);
  const u16* downw[3];
  for (int i = 0; i < 3; ++i) downw[i] = cvtW(down[i].proj.w, dims[i], dims[i + 1]);
  struct BlkW { const u16 *ak, *am1, *am2, *aoL, *aq, *av, *ffn1, *ffn2, *gate, *pk, *po, *pq, *pv; } bw[12];
  for (int t = 0; t < 12; ++t) {
    const int d = blkDim[t], mm = 4 * d;
    bw[t].ak   = cvtW(blk[t].ak.w, d, d);
    bw[t].am1  = cvtW(blk[t].am1.w, d, mm);
    bw[t].am2  = cvtW(blk[t].am2.w, mm, d);
    bw[t].aoL  = cvtW(blk[t].aoL.w, d, d);
    bw[t].aq   = cvtW(blk[t].aq.w, d, d);
    bw[t].av   = cvtW(blk[t].av.w, d, d);
    bw[t].ffn1 = cvtW(blk[t].ffn1.w, d, mm);
    bw[t].ffn2 = cvtW(blk[t].ffn2.w, mm, d);
    bw[t].gate = cvtW(blk[t].gate.w, 2 * d, d);
    bw[t].pk   = cvtW(blk[t].pk.w, d, d);
    bw[t].po   = cvtW(blk[t].po.w, d, d);
    bw[t].pq   = cvtW(blk[t].pq.w, d, d);
    bw[t].pv   = cvtW(blk[t].pv.w, d, d);
  }

  hipMemcpyAsync(cA, inCoords, sizeof(float) * 2 * 4096 * 2, hipMemcpyDeviceToDevice, stream);
  hipMemcpyAsync(tA, inTimes,  sizeof(float) * 2 * 4096,     hipMemcpyDeviceToDevice, stream);
  hipMemcpyAsync(pA, inPol,    sizeof(float) * 2 * 4096,     hipMemcpyDeviceToDevice, stream);

  // ---- embedding ----
  {
    const int total = Bb * N0;
    const int threads = total * 192;
    k_embed_feat<<<(threads + 255) / 256, 256, 0, stream>>>(
        inCoords, inTimes, inPol, ctpe.w, ctpe.b, spe.w, spe.b, poltab, hbh, total);
    gemm(stream, hbh, fus1w, fus1.b, nullptr, t1h, total, 128, 192, 1, 1);
    gemm(stream, t1h, fus2w, fus2.b, nullptr, t2, total, 64, 128, 0, 0);
    k_layernorm<<<total, 32, 0, stream>>>(t2, fus_ln.g, fus_ln.b, xb, 64, 0);
  }

  float* curC = cA; float* altC = cB;
  float* curT = tA; float* altT = tBv;
  float* curP = pA; float* altP = pB;
  int Ntok = N0;
  int bi = 0;
  for (int s = 0; s < 4; ++s) {
    const int d = dims[s], H = headsA[s], mm = 4 * d;
    const int rows = Bb * Ntok;
    k_knn<<<dim3(Ntok, Bb), 32, 0, stream>>>(curC, curT, Ntok, knnBuf);
    for (int t = 0; t < depths[s]; ++t, ++bi) {
      const Blk& Bk = blk[bi];
      const BlkW& Bw = bw[bi];
      // --- PAAA branch ---
      k_layernorm<<<rows, 32, 0, stream>>>(xb, Bk.n1.g, Bk.n1.b, t1h, d, 1);
      gemm(stream, t1h, Bw.pq, Bk.pq.b, nullptr, qh, rows, d, d, 0, 1);
      gemm(stream, t1h, Bw.pk, Bk.pk.b, nullptr, kh, rows, d, d, 0, 1);
      gemm(stream, t1h, Bw.pv, Bk.pv.b, nullptr, vh, rows, d, d, 0, 1);
      k_paaa_flash<<<dim3(Ntok / 16, H, Bb), 32, 0, stream>>>(
          qh, kh, vh, curP, Bk.polbias, t2h, Ntok, d, H);
      gemm(stream, t2h, Bw.po, Bk.po.b, xb, pa, rows, d, d, 0, 0);
      k_layernorm<<<rows, 32, 0, stream>>>(pa, Bk.n2.g, Bk.n2.b, t1h, d, 1);
      gemm(stream, t1h, Bw.ffn1, Bk.ffn1.b, nullptr, hbh, rows, mm, d, 1, 1);
      gemm(stream, hbh, Bw.ffn2, Bk.ffn2.b, pa, pa, rows, d, mm, 0, 0);
      // --- ASNA branch (from original x) ---
      k_layernorm<<<rows, 32, 0, stream>>>(xb, Bk.aln1.g, Bk.aln1.b, t1h, d, 1);
      gemm(stream, t1h, Bw.aq, Bk.aq.b, nullptr, qh, rows, d, d, 0, 1);
      gemm(stream, t1h, Bw.ak, Bk.ak.b, nullptr, kh, rows, d, d, 0, 1);
      gemm(stream, t1h, Bw.av, Bk.av.b, nullptr, vh, rows, d, d, 0, 1);
      k_asna_attn<<<dim3(Ntok, Bb), 32, 0, stream>>>(qh, kh, vh, knnBuf, t2h, Ntok, d, H);
      gemm(stream, t2h, Bw.aoL, Bk.aoL.b, xb, ao, rows, d, d, 0, 0);
      k_layernorm<<<rows, 32, 0, stream>>>(ao, Bk.aln2.g, Bk.aln2.b, t1h, d, 1);
      gemm(stream, t1h, Bw.am1, Bk.am1.b, nullptr, hbh, rows, mm, d, 1, 1);
      gemm(stream, hbh, Bw.am2, Bk.am2.b, ao, ao, rows, d, mm, 0, 0);
      // --- gate ---
      const size_t nc = (size_t)rows * 2 * d;
      k_concat<<<(unsigned)((nc + 255) / 256), 256, 0, stream>>>(pa, ao, hbh, rows, d);
      gemm(stream, hbh, Bw.gate, Bk.gate.b, nullptr, t1, rows, d, 2 * d, 0, 0);
      const size_t ng = (size_t)rows * d;
      k_gate<<<(unsigned)((ng + 255) / 256), 256, 0, stream>>>(pa, ao, t1, xb, ng);
    }
    if (s < 3) {
      const int M = Ntok / 4;
      k_fps<<<Bb, 256, 0, stream>>>(curC, curT, fpsD, fpsI, Ntok, M);
      k_gather_feats<<<dim3(M, Bb), 256, 0, stream>>>(xb, fpsI, t2, Ntok, M, d);
      k_gather_meta<<<Bb, 256, 0, stream>>>(curC, curT, curP, fpsI, altC, altT, altP, Ntok, M);
      float* tmp;
      tmp = curC; curC = altC; altC = tmp;
      tmp = curT; curT = altT; altT = tmp;
      tmp = curP; curP = altP; altP = tmp;
      const size_t ge = (size_t)Bb * M * d;
      k_cvt<<<(unsigned)((ge + 255) / 256), 256, 0, stream>>>(t2, t2h, ge);
      const int d2 = dims[s + 1];
      gemm(stream, t2h, downw[s], down[s].proj.b, nullptr, t1, Bb * M, d2, d, 0, 0);
      k_layernorm<<<Bb * M, 32, 0, stream>>>(t1, down[s].ln.g, down[s].ln.b, xb, d2, 0);
      Ntok = M;
    }
  }

  // outputs: feats [2,64,512], coords [2,64,2], times [2,64], polarities [2,64]
  float* out = (float*)d_out;
  hipMemcpyAsync(out, xb, sizeof(float) * 2 * 64 * 512, hipMemcpyDeviceToDevice, stream);
  hipMemcpyAsync(out + 2 * 64 * 512, curC, sizeof(float) * 2 * 64 * 2,
                 hipMemcpyDeviceToDevice, stream);
  hipMemcpyAsync(out + 2 * 64 * 512 + 256, curT, sizeof(float) * 2 * 64,
                 hipMemcpyDeviceToDevice, stream);
  hipMemcpyAsync(out + 2 * 64 * 512 + 256 + 128, curP, sizeof(float) * 2 * 64,
                 hipMemcpyDeviceToDevice, stream);
}
